// _LstmCellWithProjection_68418829025600
// MI455X (gfx1250) — matrix operand order repaired and checked by reading, not compiled, not run on hardware
//
#include <hip/hip_runtime.h>
#include <hip/hip_bf16.h>

// Problem constants (from reference)
#define BB 32
#define TT 128
#define INW 512
#define HH 512
#define CC 4096
#define KCAT 1024     // IN + H
#define G4 (4*CC)     // 16384 gates
#define NCT 256       // channel tiles of 16 (CC/16)
#define NKB 32        // k-blocks of 32 (KCAT/32)
#define NNT 32        // proj N tiles (HH/16)
#define NKBP 128      // proj k-blocks (CC/32)

typedef __attribute__((ext_vector_type(16))) __bf16 v16bf;
typedef __attribute__((ext_vector_type(8)))  __bf16 v8bf;
typedef __attribute__((ext_vector_type(8)))  float  v8f;

__device__ __forceinline__ __bf16 f2bf(float f) {
    union { float f; unsigned u; } v; v.f = f;
    unsigned r = v.u + 0x7FFFu + ((v.u >> 16) & 1u);   // RNE
    unsigned short h = (unsigned short)(r >> 16);
    __bf16 b; __builtin_memcpy(&b, &h, 2);
    return b;
}

__device__ __forceinline__ float sigmoidf(float x) {
    return 1.0f / (1.0f + __expf(-x));
}

#define WMMA_BF16(A, Bv, Cacc) \
    __builtin_amdgcn_wmma_f32_16x16x32_bf16(false, (A), false, (Bv), (short)0, (Cacc), false, false)

// ---------------- prep kernels ----------------

// Pack gate weights into WMMA-B-tile order:
//   dst index i = ((((q*NCT + ct)*NKB + kb)*32 + lane)*16 + e)
//   maps to gate row g = q*CC + ct*16 + (lane&15), k = kb*32 + (e&7) + 8*(lane>>4) + 16*(e>>3)
// so each wave's B load is 1KB fully contiguous.
__global__ void k_pack_wcat(const float* __restrict__ wi, const float* __restrict__ wsrc,
                            __bf16* __restrict__ dst, long n) {
    for (long i = (long)blockIdx.x * blockDim.x + threadIdx.x; i < n;
         i += (long)gridDim.x * blockDim.x) {
        const int e    = (int)(i & 15);
        const int lane = (int)((i >> 4) & 31);
        const int kb   = (int)((i >> 9) & 31);
        const int ct   = (int)((i >> 14) & 255);
        const int q    = (int)(i >> 22);
        const long g   = (long)q * CC + ct * 16 + (lane & 15);
        const int  k   = kb * 32 + (e & 7) + 8 * (lane >> 4) + 16 * (e >> 3);
        const float v  = (k < INW) ? wi[g * INW + k] : wsrc[g * HH + (k - INW)];
        dst[i] = f2bf(v);
    }
}

// Pack w_proj the same way: [nt][kb][lane][16]
__global__ void k_pack_wproj(const float* __restrict__ wp, __bf16* __restrict__ dst, long n) {
    for (long i = (long)blockIdx.x * blockDim.x + threadIdx.x; i < n;
         i += (long)gridDim.x * blockDim.x) {
        const int e    = (int)(i & 15);
        const int lane = (int)((i >> 4) & 31);
        const int kb   = (int)((i >> 9) & 127);
        const int nt   = (int)(i >> 16);
        const long nrow = (long)nt * 16 + (lane & 15);
        const int  k    = kb * 32 + (e & 7) + 8 * (lane >> 4) + 16 * (e >> 3);
        dst[i] = f2bf(wp[nrow * CC + k]);
    }
}

__global__ void k_cvt(const float* __restrict__ src, __bf16* __restrict__ dst, long n) {
    for (long i = (long)blockIdx.x * blockDim.x + threadIdx.x; i < n;
         i += (long)gridDim.x * blockDim.x)
        dst[i] = f2bf(src[i]);
}

__global__ void k_zero_state(float* __restrict__ h, __bf16* __restrict__ hbf,
                             float* __restrict__ c) {
    for (long i = (long)blockIdx.x * blockDim.x + threadIdx.x; i < (long)BB * CC;
         i += (long)gridDim.x * blockDim.x) {
        c[i] = 0.0f;
        if (i < (long)BB * HH) { h[i] = 0.0f; hbf[i] = f2bf(0.0f); }
    }
}

// ---------------- recurrent step: gates + cell update ----------------
// One wave per 16-wide channel tile: both batch halves x 4 gate types
// (8 named WMMA accumulator chains; each B tile fetched exactly once).
__global__ void __launch_bounds__(128, 1)
k_gates(const __bf16* __restrict__ x,     // (B,T,IN) bf16
        const __bf16* __restrict__ hbf,   // (B,H) bf16
        const __bf16* __restrict__ wcatp, // packed (4C x 1024) bf16
        const float*  __restrict__ bstate,// (4C) f32
        const int*    __restrict__ lens,  // (B)
        float*        __restrict__ c,     // (B,C) f32 state
        __bf16*       __restrict__ abf,   // (B,C) bf16: o*tanh(c_new)
        int t) {
    const int lane  = threadIdx.x & 31;
    const int ct    = ((int)blockIdx.x * blockDim.x + threadIdx.x) >> 5;  // 0..255
    const int halfL = lane >> 4;
    const int lmod  = lane & 15;
    const int ch    = ct * 16 + lmod;

    const __bf16* xrow0 = x   + ((size_t)(lmod)      * TT + t) * INW + 8 * halfL;
    const __bf16* xrow1 = x   + ((size_t)(16 + lmod) * TT + t) * INW + 8 * halfL;
    const __bf16* hrow0 = hbf + (size_t)(lmod)      * HH + 8 * halfL;
    const __bf16* hrow1 = hbf + (size_t)(16 + lmod) * HH + 8 * halfL;

    // Strength-reduced packed-B pointers: step 1KB (32 lanes * 32B) per k-block.
    const size_t QSTRIDE = (size_t)NCT * NKB * 32 * 16;   // elems per gate type
    const __bf16* bp0 = wcatp + (size_t)ct * NKB * 32 * 16 + (size_t)lane * 16;
    const __bf16* bp1 = bp0 + QSTRIDE;
    const __bf16* bp2 = bp1 + QSTRIDE;
    const __bf16* bp3 = bp2 + QSTRIDE;

    v8f ai0 = v8f{}, ai1 = v8f{};   // gate i, halves 0/1
    v8f af0 = v8f{}, af1 = v8f{};   // gate f
    v8f ag0 = v8f{}, ag1 = v8f{};   // gate g
    v8f ao0 = v8f{}, ao1 = v8f{};   // gate o

    for (int kb = 0; kb < NKB; ++kb) {
        const int kblock = kb * 32;
        const __bf16* a0 = (kb < 16) ? (xrow0 + kblock) : (hrow0 + (kblock - INW));
        const __bf16* a1 = (kb < 16) ? (xrow1 + kblock) : (hrow1 + (kblock - INW));
        v8bf a0lo = *(const v8bf*)(a0);
        v8bf a0hi = *(const v8bf*)(a0 + 16);
        v8bf a1lo = *(const v8bf*)(a1);
        v8bf a1hi = *(const v8bf*)(a1 + 16);
        v16bf A0, A1;
        #pragma unroll
        for (int j = 0; j < 8; ++j) {
            A0[j] = a0lo[j]; A0[j + 8] = a0hi[j];
            A1[j] = a1lo[j]; A1[j + 8] = a1hi[j];
        }

        // All four B tiles up front: lets the scheduler retire WMMAs against
        // partially drained LOADcnt instead of full waits.
        v16bf B0 = *(const v16bf*)bp0;
        v16bf B1 = *(const v16bf*)bp1;
        v16bf B2 = *(const v16bf*)bp2;
        v16bf B3 = *(const v16bf*)bp3;
        if (kb + 1 < NKB) {   // global_prefetch_b8 into L2 for next k-block
            __builtin_prefetch((const void*)(bp0 + 512), 0, 3);
            __builtin_prefetch((const void*)(bp1 + 512), 0, 3);
            __builtin_prefetch((const void*)(bp2 + 512), 0, 3);
            __builtin_prefetch((const void*)(bp3 + 512), 0, 3);
        }

        ai0 = WMMA_BF16(A0, B0, ai0);
        ai1 = WMMA_BF16(A1, B0, ai1);
        af0 = WMMA_BF16(A0, B1, af0);
        af1 = WMMA_BF16(A1, B1, af1);
        ag0 = WMMA_BF16(A0, B2, ag0);
        ag1 = WMMA_BF16(A1, B2, ag1);
        ao0 = WMMA_BF16(A0, B3, ao0);
        ao1 = WMMA_BF16(A1, B3, ao1);

        bp0 += 512; bp1 += 512; bp2 += 512; bp3 += 512;   // 32 lanes * 16 elems
    }

    // Epilogue: bias + gates + cell update (per lane: fixed ch, 16 batch rows)
    const float bi = bstate[ch];
    const float bf_ = bstate[CC + ch];
    const float bg = bstate[2 * CC + ch];
    const float bo = bstate[3 * CC + ch];

    #pragma unroll
    for (int m = 0; m < 2; ++m) {
        const v8f& Ai = m ? ai1 : ai0;
        const v8f& Af = m ? af1 : af0;
        const v8f& Ag = m ? ag1 : ag0;
        const v8f& Ao = m ? ao1 : ao0;
        #pragma unroll
        for (int r = 0; r < 8; ++r) {
            const int b = m * 16 + r + 8 * halfL;
            const size_t idx = (size_t)b * CC + ch;
            const float cold = c[idx];
            const float iv = sigmoidf(Ai[r] + bi);
            const float fv = sigmoidf(Af[r] + bf_);
            const float gv = tanhf(Ag[r] + bg);
            const float ov = sigmoidf(Ao[r] + bo);
            float cn = iv * gv + fv * cold;
            cn = fminf(fmaxf(cn, -3.0f), 3.0f);
            if (t < lens[b]) c[idx] = cn;         // masked rows keep old c
            abf[idx] = f2bf(ov * tanhf(cn));      // unused downstream when masked
        }
    }
}

// ---------------- recurrent step: projection ----------------
// h_new = clip( a (32x4096) @ w_proj^T, +-3 ); one wave per N tile, both halves.
__global__ void __launch_bounds__(128, 1)
k_proj(const __bf16* __restrict__ abf,    // (B,C) bf16
       const __bf16* __restrict__ wprojp, // packed (H x C) bf16
       const int*    __restrict__ lens,
       float*        __restrict__ h,      // (B,H) f32 state
       __bf16*       __restrict__ hbf,    // (B,H) bf16 state
       float*        __restrict__ out,    // (B,T,H) f32
       int t) {
    const int lane  = threadIdx.x & 31;
    const int nt    = ((int)blockIdx.x * blockDim.x + threadIdx.x) >> 5;  // 0..31
    const int halfL = lane >> 4;
    const int lmod  = lane & 15;
    const int n     = nt * 16 + lmod;

    const __bf16* arow0 = abf + (size_t)(lmod)      * CC + 8 * halfL;
    const __bf16* arow1 = abf + (size_t)(16 + lmod) * CC + 8 * halfL;
    const __bf16* bp    = wprojp + (size_t)nt * NKBP * 32 * 16 + (size_t)lane * 16;

    v8f acc0 = v8f{}, acc1 = v8f{};

    for (int kb = 0; kb < NKBP; ++kb) {
        const int kblock = kb * 32;
        v8bf a0lo = *(const v8bf*)(arow0 + kblock);
        v8bf a0hi = *(const v8bf*)(arow0 + kblock + 16);
        v8bf a1lo = *(const v8bf*)(arow1 + kblock);
        v8bf a1hi = *(const v8bf*)(arow1 + kblock + 16);
        v16bf A0, A1;
        #pragma unroll
        for (int j = 0; j < 8; ++j) {
            A0[j] = a0lo[j]; A0[j + 8] = a0hi[j];
            A1[j] = a1lo[j]; A1[j + 8] = a1hi[j];
        }

        v16bf Bv = *(const v16bf*)bp;
        if (kb + 1 < NKBP)
            __builtin_prefetch((const void*)(bp + 512), 0, 3);

        acc0 = WMMA_BF16(A0, Bv, acc0);
        acc1 = WMMA_BF16(A1, Bv, acc1);
        bp += 512;
    }

    #pragma unroll
    for (int m = 0; m < 2; ++m) {
        const v8f& acc = m ? acc1 : acc0;
        #pragma unroll
        for (int r = 0; r < 8; ++r) {
            const int b = m * 16 + r + 8 * halfL;
            float v = fminf(fmaxf(acc[r], -3.0f), 3.0f);
            const size_t oidx = ((size_t)b * TT + t) * HH + n;
            if (t < lens[b]) {
                out[oidx] = v;
                h[(size_t)b * HH + n] = v;
                hbf[(size_t)b * HH + n] = f2bf(v);
            } else {
                out[oidx] = 0.0f;                 // reference emits zeros past length
            }
        }
    }
}

// ---------------- final state emit ----------------
__global__ void k_final(const float* __restrict__ h, const float* __restrict__ c,
                        float* __restrict__ dst) {
    const long nh = (long)BB * HH, nc = (long)BB * CC;
    for (long i = (long)blockIdx.x * blockDim.x + threadIdx.x; i < nh + nc;
         i += (long)gridDim.x * blockDim.x)
        dst[i] = (i < nh) ? h[i] : c[i - nh];
}

extern "C" void kernel_launch(void* const* d_in, const int* in_sizes, int n_in,
                              void* d_out, int out_size, void* d_ws, size_t ws_size,
                              hipStream_t stream) {
    const float* x_f     = (const float*)d_in[0];   // (B,T,IN)
    const int*   lens    = (const int*)d_in[1];     // (B,)
    const float* w_input = (const float*)d_in[2];   // (4C,IN)
    const float* w_state = (const float*)d_in[3];   // (4C,H)
    const float* b_state = (const float*)d_in[4];   // (4C,)
    const float* w_proj  = (const float*)d_in[5];   // (H,C)
    float* out = (float*)d_out;

    char* ws = (char*)d_ws;
    size_t off = 0;
    __bf16* wcatp  = (__bf16*)(ws + off); off += (size_t)G4 * KCAT * 2;      // 33.5 MB
    __bf16* wprojp = (__bf16*)(ws + off); off += (size_t)HH * CC * 2;        // 4 MB
    __bf16* xb     = (__bf16*)(ws + off); off += (size_t)BB * TT * INW * 2;  // 4 MB
    float*  hst    = (float*)(ws + off);  off += (size_t)BB * HH * 4;
    __bf16* hbf    = (__bf16*)(ws + off); off += (size_t)BB * HH * 2;
    float*  cst    = (float*)(ws + off);  off += (size_t)BB * CC * 4;
    __bf16* ab     = (__bf16*)(ws + off); off += (size_t)BB * CC * 2;

    // One-time prep (every launch: deterministic, no caching)
    k_pack_wcat<<<8192, 256, 0, stream>>>(w_input, w_state, wcatp, (long)G4 * KCAT);
    k_pack_wproj<<<2048, 256, 0, stream>>>(w_proj, wprojp, (long)HH * CC);
    k_cvt<<<4096, 256, 0, stream>>>(x_f, xb, (long)BB * TT * INW);
    k_zero_state<<<512, 256, 0, stream>>>(hst, hbf, cst);

    // Sequential scan over time (stream order carries the dependency)
    for (int t = 0; t < TT; ++t) {
        k_gates<<<64, 128, 0, stream>>>(xb, hbf, wcatp, b_state, lens, cst, ab, t);
        k_proj<<<8, 128, 0, stream>>>(ab, wprojp, lens, hst, hbf, out, t);
    }

    // Final state tail: [h (B*H) | c (B*C)] after output accumulator
    k_final<<<128, 256, 0, stream>>>(hst, cst, out + (size_t)BB * TT * HH);
}
